// QuantizedLinear_44315472560502
// MI455X (gfx1250) — matrix operand order repaired and checked by reading, not compiled, not run on hardware
//
#include <hip/hip_runtime.h>
#include <cstdint>

// ---------------------------------------------------------------------------
// QuantizedLinear on gfx1250: y = x @ dequant(Wq)^T + dequant(bq)
//   M=8192 (B*S), N=11008 (OUT_F), K=4096 (IN_F)
// Fast path: x -> fp16 and W -> fp16 once (workspace), then WMMA GEMM whose
// inner loop is GLOBAL_LOAD_ASYNC_TO_LDS_B128 staging + ds_load_b128 +
// v_wmma_f32_16x16x32_f16 at a 1:1 ds:wmma ratio (64x64 wave tile).
// ---------------------------------------------------------------------------

typedef __attribute__((ext_vector_type(16))) _Float16 v16h;
typedef __attribute__((ext_vector_type(8)))  _Float16 v8h;
typedef __attribute__((ext_vector_type(8)))  float    v8f;
typedef __attribute__((ext_vector_type(4)))  float    v4f;
typedef __attribute__((ext_vector_type(4)))  int      v4i;

#define M_DIM 8192
#define N_DIM 11008
#define K_DIM 4096

// ---- CDNA5 async-to-LDS path (guarded; sync register-staging fallback) ----
#if defined(__has_builtin)
#if __has_builtin(__builtin_amdgcn_global_load_async_to_lds_b128)
#define HAVE_ASYNC_LDS 1
#endif
#endif

#if defined(__has_builtin)
#if __has_builtin(__builtin_amdgcn_s_wait_asynccnt)
#define WAIT_ASYNC(n) __builtin_amdgcn_s_wait_asynccnt(n)
#else
#define WAIT_ASYNC(n) asm volatile("s_wait_asynccnt %0" ::"i"(n) : "memory")
#endif
#else
#define WAIT_ASYNC(n) asm volatile("s_wait_asynccnt %0" ::"i"(n) : "memory")
#endif

#if HAVE_ASYNC_LDS
// Builtin signature (from clang diagnostic): (global int4*, local int4*, Ii, Ii)
typedef __attribute__((address_space(1))) v4i* g_v4i_p;
typedef __attribute__((address_space(3))) v4i* l_v4i_p;

__device__ __forceinline__ g_v4i_p as_global_v4i(const void* p) {
  return (g_v4i_p)(v4i*)const_cast<void*>(p);
}
__device__ __forceinline__ l_v4i_p as_local_v4i(void* p) {
  return (l_v4i_p)(v4i*)p;
}
#endif

// ------------------------- x fp32 -> fp16 ----------------------------------
__global__ __launch_bounds__(256) void x_to_half_kernel(
    const float* __restrict__ x, _Float16* __restrict__ xh, long long n) {
  long long i = ((long long)blockIdx.x * blockDim.x + threadIdx.x) * 8;
  if (i + 8 <= n) {
    v4f f0 = *(const v4f*)(x + i);
    v4f f1 = *(const v4f*)(x + i + 4);
    v8h h;
#pragma unroll
    for (int j = 0; j < 4; ++j) {
      h[j]     = (_Float16)f0[j];
      h[4 + j] = (_Float16)f1[j];
    }
    *(v8h*)(xh + i) = h;
  }
}

// ------------------------- Wq int8 -> fp16 dequant -------------------------
__global__ __launch_bounds__(256) void w_to_half_kernel(
    const int8_t* __restrict__ wq, _Float16* __restrict__ wh,
    const float* __restrict__ wsp, const float* __restrict__ wzpp,
    long long n) {
  long long i = ((long long)blockIdx.x * blockDim.x + threadIdx.x) * 16;
  if (i + 16 <= n) {
    const float ws = *wsp, wzp = *wzpp;
    v4i q = *(const v4i*)(wq + i);
    const int8_t* qb = (const int8_t*)&q;
    v8h h0, h1;
#pragma unroll
    for (int j = 0; j < 8; ++j) {
      h0[j] = (_Float16)(ws * ((float)qb[j]     - wzp));
      h1[j] = (_Float16)(ws * ((float)qb[8 + j] - wzp));
    }
    *(v8h*)(wh + i)     = h0;
    *(v8h*)(wh + i + 8) = h1;
  }
}

// ------------------------------ fast GEMM ----------------------------------
// BM=128, BN=256, BK=32; 8 waves, each computing a 64x64 patch (4x4 WMMA
// tiles): 16 ds_load_b128 feed 16 v_wmma per k-step (1:1). LDS 48 KB.
#define FBM 128
#define FBN 256
#define FBK 32
#define FNK (K_DIM / FBK)  // 128

__global__ __launch_bounds__(256) void qlinear_wmma_fast_kernel(
    const _Float16* __restrict__ Ah,   // [M,K]
    const _Float16* __restrict__ Wh,   // [N,K]
    const int8_t* __restrict__ Bq,     // [N]
    const float* __restrict__ bscale_p, const float* __restrict__ bzp_p,
    float* __restrict__ out)           // [M,N]
{
  __shared__ _Float16 As[2][FBM][FBK];  // 16 KB
  __shared__ _Float16 Bs[2][FBN][FBK];  // 32 KB

  const int tid   = threadIdx.x;
  const int lane  = tid & 31;
  const int wave  = tid >> 5;
  const int waveM = wave >> 2;   // 0..1 -> 64-row slab
  const int waveN = wave & 3;    // 0..3 -> 64-col slab
  const int lrow  = lane & 15;
  const bool hi   = lane >= 16;

  const int mblk = blockIdx.y * FBM;
  const int nblk = blockIdx.x * FBN;

  // Staging: rows are 32 halves = 64 B = 4 chunks of 16 B.
  // A tile: 512 chunks (2/thread); B tile: 1024 chunks (4/thread).
#if HAVE_ASYNC_LDS
  auto issue_tile = [&](int kk, int buf) {
    const size_t k0 = (size_t)kk * FBK;
#pragma unroll
    for (int i = 0; i < 2; ++i) {
      const int c = tid + i * 256;
      const int r = c >> 2, cl = (c & 3) * 8;
      __builtin_amdgcn_global_load_async_to_lds_b128(
          as_global_v4i(Ah + (size_t)(mblk + r) * K_DIM + k0 + cl),
          as_local_v4i(&As[buf][r][cl]), 0, 0);
    }
#pragma unroll
    for (int i = 0; i < 4; ++i) {
      const int c = tid + i * 256;
      const int r = c >> 2, cl = (c & 3) * 8;
      __builtin_amdgcn_global_load_async_to_lds_b128(
          as_global_v4i(Wh + (size_t)(nblk + r) * K_DIM + k0 + cl),
          as_local_v4i(&Bs[buf][r][cl]), 0, 0);
    }
  };
#else
  v8h ar[2], br[4];
  auto g_load = [&](int kk) {
    const size_t k0 = (size_t)kk * FBK;
#pragma unroll
    for (int i = 0; i < 2; ++i) {
      const int c = tid + i * 256;
      const int r = c >> 2, cl = (c & 3) * 8;
      ar[i] = *(const v8h*)(Ah + (size_t)(mblk + r) * K_DIM + k0 + cl);
    }
#pragma unroll
    for (int i = 0; i < 4; ++i) {
      const int c = tid + i * 256;
      const int r = c >> 2, cl = (c & 3) * 8;
      br[i] = *(const v8h*)(Wh + (size_t)(nblk + r) * K_DIM + k0 + cl);
    }
  };
  auto s_store = [&](int buf) {
#pragma unroll
    for (int i = 0; i < 2; ++i) {
      const int c = tid + i * 256;
      const int r = c >> 2, cl = (c & 3) * 8;
      *(v8h*)&As[buf][r][cl] = ar[i];
    }
#pragma unroll
    for (int i = 0; i < 4; ++i) {
      const int c = tid + i * 256;
      const int r = c >> 2, cl = (c & 3) * 8;
      *(v8h*)&Bs[buf][r][cl] = br[i];
    }
  };
#endif

  v8f acc[4][4];
#pragma unroll
  for (int mt = 0; mt < 4; ++mt)
#pragma unroll
    for (int nt = 0; nt < 4; ++nt)
#pragma unroll
      for (int r = 0; r < 8; ++r) acc[mt][nt][r] = 0.0f;

#if HAVE_ASYNC_LDS
  issue_tile(0, 0);
  WAIT_ASYNC(0);
  __syncthreads();
#else
  g_load(0);
  s_store(0);
  __syncthreads();
#endif

  for (int kk = 0; kk < FNK; ++kk) {
    const int cur = kk & 1;
#if HAVE_ASYNC_LDS
    if (kk + 1 < FNK) issue_tile(kk + 1, cur ^ 1);
#else
    if (kk + 1 < FNK) g_load(kk + 1);
#endif

    // ---- fragments: A lanes 0-15 hold K 0-7 & 16-23 (hi: 8-15 & 24-31);
    //      B lane holds K 0-7 & 16-23 of column N=lane%16 (hi: 8-15 & 24-31)
    v16h afrag[4];
    const int aoff = hi ? 8 : 0;
#pragma unroll
    for (int mt = 0; mt < 4; ++mt) {
      const _Float16* p = &As[cur][waveM * 64 + mt * 16 + lrow][aoff];
      v8h ca = *(const v8h*)p;
      v8h cb = *(const v8h*)(p + 16);
#pragma unroll
      for (int j = 0; j < 8; ++j) {
        afrag[mt][j]     = ca[j];
        afrag[mt][8 + j] = cb[j];
      }
    }
    v16h bfrag[4];
    const int boff = hi ? 8 : 0;
#pragma unroll
    for (int nt = 0; nt < 4; ++nt) {
      const _Float16* p = &Bs[cur][waveN * 64 + nt * 16 + lrow][boff];
      v8h ca = *(const v8h*)p;
      v8h cb = *(const v8h*)(p + 16);
#pragma unroll
      for (int j = 0; j < 8; ++j) {
        bfrag[nt][j]     = ca[j];
        bfrag[nt][8 + j] = cb[j];
      }
    }
#pragma unroll
    for (int mt = 0; mt < 4; ++mt)
#pragma unroll
      for (int nt = 0; nt < 4; ++nt)
        acc[mt][nt] = __builtin_amdgcn_wmma_f32_16x16x32_f16(
            false, afrag[mt], false, bfrag[nt],
            (short)0, acc[mt][nt], false, false);

#if HAVE_ASYNC_LDS
    if (kk + 1 < FNK) {
      WAIT_ASYNC(6);   // previous tile's 6 issues done; next tile may fly
      __syncthreads();
    } else {
      __syncthreads();
    }
#else
    __syncthreads();
    if (kk + 1 < FNK) s_store(cur ^ 1);
    __syncthreads();
#endif
  }

  const float bs  = *bscale_p;
  const float bzp = *bzp_p;
#pragma unroll
  for (int nt = 0; nt < 4; ++nt) {
    const int col = nblk + waveN * 64 + nt * 16 + lrow;
    const float bv = bs * ((float)Bq[col] - bzp);
#pragma unroll
    for (int mt = 0; mt < 4; ++mt) {
      const int rbase = mblk + waveM * 64 + mt * 16 + (hi ? 8 : 0);
#pragma unroll
      for (int r = 0; r < 8; ++r)
        out[(size_t)(rbase + r) * N_DIM + col] = acc[mt][nt][r] + bv;
    }
  }
}

// ---------------- fallback GEMM (in-kernel weight dequant, BK=32) ----------
#define BM 128
#define BN 128
#define BK 32
#define NK (K_DIM / BK)

template <bool A_HALF>
__global__ __launch_bounds__(256) void qlinear_wmma_kernel(
    const void* __restrict__ Aptr,
    const int8_t* __restrict__ Wq, const int8_t* __restrict__ Bq,
    const float* __restrict__ wscale_p, const float* __restrict__ wzp_p,
    const float* __restrict__ bscale_p, const float* __restrict__ bzp_p,
    float* __restrict__ out) {
  __shared__ _Float16 As[2][BM][BK];
  __shared__ _Float16 Bs[2][BN][BK];

  const int tid = threadIdx.x, lane = tid & 31, wave = tid >> 5;
  const int waveM = wave >> 2, waveN = wave & 3;
  const int lrow = lane & 15;
  const bool hi = lane >= 16;
  const int mblk = blockIdx.y * BM, nblk = blockIdx.x * BN;
  const float ws = *wscale_p, wzp = *wzp_p;

  const int srow = tid >> 1, scol = (tid & 1) * 16;
  const _Float16* Ah = (const _Float16*)Aptr;
  const float* Af = (const float*)Aptr;
  const size_t a_goff = (size_t)(mblk + srow) * K_DIM + scol;
  const size_t b_goff = (size_t)(nblk + srow) * K_DIM + scol;

  v8h a_r0, a_r1;
  v4i bq_r;
  auto g_load = [&](int kk) {
    const size_t k0 = (size_t)kk * BK;
    if (A_HALF) {
      const _Float16* p = Ah + a_goff + k0;
      a_r0 = *(const v8h*)p;
      a_r1 = *(const v8h*)(p + 8);
    } else {
      const float* p = Af + a_goff + k0;
      v4f f0 = *(const v4f*)(p + 0), f1 = *(const v4f*)(p + 4);
      v4f f2 = *(const v4f*)(p + 8), f3 = *(const v4f*)(p + 12);
#pragma unroll
      for (int j = 0; j < 4; ++j) {
        a_r0[j] = (_Float16)f0[j]; a_r0[4 + j] = (_Float16)f1[j];
        a_r1[j] = (_Float16)f2[j]; a_r1[4 + j] = (_Float16)f3[j];
      }
    }
    bq_r = *(const v4i*)(Wq + b_goff + k0);
  };
  auto s_store = [&](int buf) {
    *(v8h*)&As[buf][srow][scol] = a_r0;
    *(v8h*)&As[buf][srow][scol + 8] = a_r1;
    const int8_t* q = (const int8_t*)&bq_r;
    v8h w0, w1;
#pragma unroll
    for (int j = 0; j < 8; ++j) {
      w0[j] = (_Float16)(ws * ((float)q[j] - wzp));
      w1[j] = (_Float16)(ws * ((float)q[8 + j] - wzp));
    }
    *(v8h*)&Bs[buf][srow][scol] = w0;
    *(v8h*)&Bs[buf][srow][scol + 8] = w1;
  };

  v8f acc[4][2];
#pragma unroll
  for (int mt = 0; mt < 4; ++mt)
#pragma unroll
    for (int nt = 0; nt < 2; ++nt)
#pragma unroll
      for (int r = 0; r < 8; ++r) acc[mt][nt][r] = 0.0f;

  g_load(0);
  s_store(0);
  __syncthreads();

  for (int kk = 0; kk < NK; ++kk) {
    const int cur = kk & 1;
    if (kk + 1 < NK) g_load(kk + 1);
    v16h afrag[4];
    const int aoff = hi ? 8 : 0;
#pragma unroll
    for (int mt = 0; mt < 4; ++mt) {
      const _Float16* p = &As[cur][waveM * 64 + mt * 16 + lrow][aoff];
      v8h ca = *(const v8h*)p;
      v8h cb = *(const v8h*)(p + 16);
#pragma unroll
      for (int j = 0; j < 8; ++j) { afrag[mt][j] = ca[j]; afrag[mt][8 + j] = cb[j]; }
    }
    v16h bfrag[2];
    const int boff = hi ? 8 : 0;
#pragma unroll
    for (int nt = 0; nt < 2; ++nt) {
      const _Float16* p = &Bs[cur][waveN * 32 + nt * 16 + lrow][boff];
      v8h ca = *(const v8h*)p;
      v8h cb = *(const v8h*)(p + 16);
#pragma unroll
      for (int j = 0; j < 8; ++j) { bfrag[nt][j] = ca[j]; bfrag[nt][8 + j] = cb[j]; }
    }
#pragma unroll
    for (int mt = 0; mt < 4; ++mt)
#pragma unroll
      for (int nt = 0; nt < 2; ++nt)
        acc[mt][nt] = __builtin_amdgcn_wmma_f32_16x16x32_f16(
            false, afrag[mt], false, bfrag[nt], (short)0, acc[mt][nt], false, false);
    __syncthreads();
    if (kk + 1 < NK) s_store(cur ^ 1);
    __syncthreads();
  }

  const float bs = *bscale_p, bzp = *bzp_p;
#pragma unroll
  for (int nt = 0; nt < 2; ++nt) {
    const int col = nblk + waveN * 32 + nt * 16 + lrow;
    const float bv = bs * ((float)Bq[col] - bzp);
#pragma unroll
    for (int mt = 0; mt < 4; ++mt) {
      const int rbase = mblk + waveM * 64 + mt * 16 + (hi ? 8 : 0);
#pragma unroll
      for (int r = 0; r < 8; ++r)
        out[(size_t)(rbase + r) * N_DIM + col] = acc[mt][nt][r] + bv;
    }
  }
}

// ---------------------------------------------------------------------------
extern "C" void kernel_launch(void* const* d_in, const int* in_sizes, int n_in,
                              void* d_out, int out_size, void* d_ws, size_t ws_size,
                              hipStream_t stream) {
  const float*  x     = (const float*)d_in[0];
  const int8_t* wq    = (const int8_t*)d_in[1];
  const int8_t* bq    = (const int8_t*)d_in[2];
  const float*  wscal = (const float*)d_in[3];
  const float*  wzp   = (const float*)d_in[4];
  const float*  bscal = (const float*)d_in[5];
  const float*  bzp   = (const float*)d_in[6];
  float* out = (float*)d_out;

  const long long nx = (long long)M_DIM * K_DIM;  // 33.5M
  const long long nw = (long long)N_DIM * K_DIM;  // 45.1M
  const size_t need_x  = (size_t)nx * sizeof(_Float16);            // 67 MB
  const size_t need_xw = need_x + (size_t)nw * sizeof(_Float16);   // 157 MB

  if (ws_size >= need_xw) {
    _Float16* xh = (_Float16*)d_ws;
    _Float16* wh = (_Float16*)((char*)d_ws + need_x);
    x_to_half_kernel<<<(unsigned)((nx / 8 + 255) / 256), 256, 0, stream>>>(x, xh, nx);
    w_to_half_kernel<<<(unsigned)((nw / 16 + 255) / 256), 256, 0, stream>>>(wq, wh, wscal, wzp, nw);
    dim3 fgrid(N_DIM / FBN, M_DIM / FBM);  // 43 x 64
    qlinear_wmma_fast_kernel<<<fgrid, 256, 0, stream>>>(xh, wh, bq, bscal, bzp, out);
  } else if (ws_size >= need_x) {
    _Float16* xh = (_Float16*)d_ws;
    x_to_half_kernel<<<(unsigned)((nx / 8 + 255) / 256), 256, 0, stream>>>(x, xh, nx);
    dim3 grid(N_DIM / BN, M_DIM / BM);     // 86 x 64
    qlinear_wmma_kernel<true><<<grid, 256, 0, stream>>>(
        xh, wq, bq, wscal, wzp, bscal, bzp, out);
  } else {
    dim3 grid(N_DIM / BN, M_DIM / BM);     // 86 x 64
    qlinear_wmma_kernel<false><<<grid, 256, 0, stream>>>(
        x, wq, bq, wscal, wzp, bscal, bzp, out);
  }
}
